// FeatureExtractorViT_50835232915590
// MI455X (gfx1250) — matrix operand order repaired and checked by reading, not compiled, not run on hardware
//
#include <hip/hip_runtime.h>
#include <hip/hip_bf16.h>

// Problem constants
#define BB   512
#define CC   3
#define SS   36
#define DD   768
#define HH   12
#define HDD  64
#define LL   4
#define OUTN 256
#define TT   (BB*CC*SS)   // 55296 token rows

typedef _Float16 v16h __attribute__((ext_vector_type(16)));
typedef _Float16 h8   __attribute__((ext_vector_type(8)));
typedef float    v8f  __attribute__((ext_vector_type(8)));
typedef unsigned int u32x4 __attribute__((ext_vector_type(4)));
typedef int          i32x4 __attribute__((ext_vector_type(4)));
typedef int          i32x8 __attribute__((ext_vector_type(8)));

#define TDM_OK __has_builtin(__builtin_amdgcn_tensor_load_to_lds)

// ---------------------------------------------------------------------------
// WMMA helpers (gfx1250, wave32). Layouts per CDNA5 ISA 7.12.2:
//  A (16x32 f16): lane m=(l&15); VGPRs hold K chunks [h*8 .. h*8+7] and
//                 [16+h*8 .. 16+h*8+7] where h = l>>4.
//  B (32x16 f16): lane n=(l&15); K chunks [h*8 .. h*8+7] and [16+h*8 .. 16+h*8+7], same as A.
//                 For X @ W^T with row-major W (NxK), B[k][n] = W[n][k] ->
//                 contiguous row chunk of W.
// ---------------------------------------------------------------------------
__device__ inline v8f wmma32(v16h a, v16h b, v8f c) {
  return __builtin_amdgcn_wmma_f32_16x16x32_f16(false, a, false, b, (short)0, c,
                                                false, false);
}

__device__ inline v16h lda_op(const _Float16* base, int m0, int k0, int ld, int lane) {
  int m  = m0 + (lane & 15);
  int kb = k0 + ((lane >> 4) << 3);
  const h8* p0 = (const h8*)(base + m * ld + kb);
  const h8* p1 = (const h8*)(base + m * ld + kb + 16);
  h8 a = *p0, b = *p1;
  v16h r;
#pragma unroll
  for (int i = 0; i < 8; ++i) { r[i] = a[i]; r[8 + i] = b[i]; }
  return r;
}

__device__ inline v16h ldb_op(const _Float16* base, int n0, int k0, int ld, int lane) {
  int n  = n0 + (lane & 15);
  int kb = k0 + ((lane >> 4) << 3);
  const h8* p0 = (const h8*)(base + n * ld + kb);
  const h8* p1 = (const h8*)(base + n * ld + kb + 16);
  h8 a = *p0, b = *p1;
  v16h r;
#pragma unroll
  for (int i = 0; i < 8; ++i) { r[i] = a[i]; r[8 + i] = b[i]; }
  return r;
}

// ---------------------------------------------------------------------------
// TDM: async DMA of a 128-row x 32-f16 tile (row stride K elements) into LDS.
// pad_enable inserts 4 DWORDs (16B) after every 16 DWORDs (64B = one 32-f16
// row), so the LDS image has stride 40 f16 = our conflict-padded layout.
// Issued by one wave; completion via s_wait_tensorcnt + barrier.
// ---------------------------------------------------------------------------
__device__ inline void tdm_tile_load(void* lds_dst, const _Float16* gsrc, int K) {
  unsigned long long ga = (unsigned long long)(const void*)gsrc;
  u32x4 g0;
  g0[0] = 1u;                                              // count=1, no gather
  g0[1] = (unsigned)(unsigned long long)lds_dst;           // LDS byte address
  g0[2] = (unsigned)ga;                                    // global addr lo
  g0[3] = ((unsigned)(ga >> 32) & 0x01FFFFFFu) | (2u << 30); // addr hi | type=2
  i32x8 g1;
  g1[0] = (1 << 16) | (1 << 20) | (3 << 22) | (3 << 25);   // 2B elems, pad 4dw/16dw
  g1[1] = (K & 0xFFFF) << 16;                              // tensor_dim0 lo16
  g1[2] = ((K >> 16) & 0xFFFF) | (128 << 16);              // dim0 hi | tensor_dim1=128
  g1[3] = (32 << 16);                                      // dim1 hi | tile_dim0=32
  g1[4] = 128;                                             // tile_dim1=128, tile_dim2=0
  g1[5] = K;                                               // tensor_dim0_stride lo32
  g1[6] = 0;
  g1[7] = 0;
  i32x4 z4 = {0, 0, 0, 0};
#if TDM_OK
 #if __clang_major__ >= 23
  i32x8 z8 = {0, 0, 0, 0, 0, 0, 0, 0};
  __builtin_amdgcn_tensor_load_to_lds(g0, g1, z4, z4, z8, 0);
 #else
  __builtin_amdgcn_tensor_load_to_lds(g0, g1, z4, z4, 0);
 #endif
#else
  (void)g0; (void)g1; (void)z4;
#endif
}

__device__ inline void stage_fence() {
#if TDM_OK
  __builtin_amdgcn_s_wait_tensorcnt(0);
#endif
}

// Stage one 128x32 K-chunk of A and W into LDS buffers (TDM or cooperative).
__device__ inline void stage(_Float16 (*Asb)[40], _Float16 (*Wsb)[40],
                             const _Float16* Abase, const _Float16* Wbase,
                             int kc, int K, int tid, int wave) {
#if TDM_OK
  if (wave == 0) {
    tdm_tile_load(&Asb[0][0], Abase + kc, K);
    tdm_tile_load(&Wsb[0][0], Wbase + kc, K);
  }
  (void)tid;
#else
  int lrow = tid >> 1, lhalf = (tid & 1) * 16;
  const _Float16* ag = Abase + (long)lrow * K + kc + lhalf;
  const _Float16* wg = Wbase + (long)lrow * K + kc + lhalf;
  *(h8*)&Asb[lrow][lhalf]     = *(const h8*)ag;
  *(h8*)&Asb[lrow][lhalf + 8] = *(const h8*)(ag + 8);
  *(h8*)&Wsb[lrow][lhalf]     = *(const h8*)wg;
  *(h8*)&Wsb[lrow][lhalf + 8] = *(const h8*)(wg + 8);
  (void)wave;
#endif
}

// ---------------------------------------------------------------------------
// Weight conversion kernels (run once per launch; deterministic)
// ---------------------------------------------------------------------------
__global__ void k_cvt_f16(const float* __restrict__ s, _Float16* __restrict__ d, int n) {
  int i = blockIdx.x * 256 + threadIdx.x;
  if (i < n) d[i] = (_Float16)s[i];
}

// W_patch (768x36) -> padded (768x64) with zero K pad
__global__ void k_cvt_wpatch(const float* __restrict__ s, _Float16* __restrict__ d) {
  int i = blockIdx.x * 256 + threadIdx.x;
  if (i >= DD * 64) return;
  int r = i >> 6, c = i & 63;
  d[i] = (c < 36) ? (_Float16)s[r * 36 + c] : (_Float16)0.0f;
}

// Wq/Wk/Wv (L,H,64,64) -> packed [L][3][H][64*64]
__global__ void k_cvt_qkv(const float* __restrict__ s, _Float16* __restrict__ d, int mat) {
  int i = blockIdx.x * 256 + threadIdx.x;
  const int n = LL * HH * 4096;
  if (i >= n) return;
  int lh = i >> 12, r = i & 4095;
  int l = lh / HH, h = lh % HH;
  d[(((l * 3 + mat) * HH) + h) * 4096 + r] = (_Float16)s[i];
}

// ---------------------------------------------------------------------------
// Patchify: (B,C,36,36) -> f16 A (55296 x 64), cols 36..63 zero
// ---------------------------------------------------------------------------
__global__ void k_patchify(const float* __restrict__ batch, _Float16* __restrict__ Ap) {
  int t = blockIdx.x;
  int j = threadIdx.x;
  int bc = t / SS, s = t % SS;
  _Float16 v = (_Float16)0.0f;
  if (j < 36) {
    int py = s / 6, px = s % 6;
    int iy = j / 6, ix = j % 6;
    v = (_Float16)batch[(long)bc * 1296 + (py * 6 + iy) * 36 + (px * 6 + ix)];
  }
  Ap[(long)t * 64 + j] = v;
}

// ---------------------------------------------------------------------------
// LayerNorm over D=768 per token. Writes f16 (always) and optionally f32.
// In-place safe for outf == x.
// ---------------------------------------------------------------------------
__global__ __launch_bounds__(256) void k_ln(const float* __restrict__ x,
                                            const float* __restrict__ g,
                                            const float* __restrict__ b,
                                            float* outf, _Float16* __restrict__ outh) {
  long row = blockIdx.x;
  const float* xr = x + row * DD;
  int tid = threadIdx.x;
  float v[3], s = 0.0f;
#pragma unroll
  for (int i = 0; i < 3; ++i) { v[i] = xr[tid + 256 * i]; s += v[i]; }
  __shared__ float red[256];
  red[tid] = s;
  __syncthreads();
  for (int off = 128; off > 0; off >>= 1) {
    if (tid < off) red[tid] += red[tid + off];
    __syncthreads();
  }
  float mean = red[0] / (float)DD;
  __syncthreads();
  float s2 = 0.0f;
#pragma unroll
  for (int i = 0; i < 3; ++i) { float d = v[i] - mean; s2 += d * d; }
  red[tid] = s2;
  __syncthreads();
  for (int off = 128; off > 0; off >>= 1) {
    if (tid < off) red[tid] += red[tid + off];
    __syncthreads();
  }
  float inv = rsqrtf(red[0] / (float)DD + 1e-5f);
#pragma unroll
  for (int i = 0; i < 3; ++i) {
    int c = tid + 256 * i;
    float y = (v[i] - mean) * inv * g[c] + b[c];
    if (outf) outf[row * DD + c] = y;
    outh[row * DD + c] = (_Float16)y;
  }
}

// ---------------------------------------------------------------------------
// Tiled WMMA GEMM:  C[M,N] = epi( A[M,K] @ W[N,K]^T + bias )
// Block tile 128x128, 256 threads / 8 waves; each wave owns a 32x64 sub-tile
// (8 f32 accumulators). K staged in 32-wide chunks, double-buffered LDS,
// moved by the Tensor Data Mover (one issuing wave) when available.
// Epilogue: +pos (row%36), +resid (f32), ReLU; writes f32 and/or f16.
// ---------------------------------------------------------------------------
__global__ __launch_bounds__(256) void k_gemm(const _Float16* __restrict__ A,
                                              const _Float16* __restrict__ W,
                                              const float* __restrict__ bias,
                                              const float* __restrict__ pos,
                                              const float* __restrict__ resid,
                                              float* Cf, _Float16* Ch,
                                              int N, int K, int relu) {
  __shared__ _Float16 As[2][128][40];   // stride 40 f16 = 80B rows (16B aligned)
  __shared__ _Float16 Ws[2][128][40];
  int tid = threadIdx.x, wave = tid >> 5, lane = tid & 31;
  int wm = wave & 3, wn = wave >> 2;    // 4 M-strips x 2 N-strips of waves
  long row0 = (long)blockIdx.x * 128;
  int  col0 = blockIdx.y * 128;

  const _Float16* Abase = A + row0 * (long)K;
  const _Float16* Wbase = W + (long)col0 * K;

  v8f acc[2][4];
#pragma unroll
  for (int mt = 0; mt < 2; ++mt)
#pragma unroll
    for (int nt = 0; nt < 4; ++nt)
#pragma unroll
      for (int i = 0; i < 8; ++i) acc[mt][nt][i] = 0.0f;

  stage(As[0], Ws[0], Abase, Wbase, 0, K, tid, wave);
  stage_fence();
  __syncthreads();

  int cur = 0;
  for (int kc = 0; kc < K; kc += 32) {
    if (kc + 32 < K)
      stage(As[cur ^ 1], Ws[cur ^ 1], Abase, Wbase, kc + 32, K, tid, wave);
    v16h a0 = lda_op(&As[cur][0][0], wm * 32,      0, 40, lane);
    v16h a1 = lda_op(&As[cur][0][0], wm * 32 + 16, 0, 40, lane);
#pragma unroll
    for (int nt = 0; nt < 4; ++nt) {
      v16h bv = ldb_op(&Ws[cur][0][0], wn * 64 + nt * 16, 0, 40, lane);
      acc[0][nt] = wmma32(a0, bv, acc[0][nt]);
      acc[1][nt] = wmma32(a1, bv, acc[1][nt]);
    }
    stage_fence();
    __syncthreads();
    cur ^= 1;
  }

#pragma unroll
  for (int mt = 0; mt < 2; ++mt) {
#pragma unroll
    for (int nt = 0; nt < 4; ++nt) {
#pragma unroll
      for (int r = 0; r < 8; ++r) {
        long m = row0 + wm * 32 + mt * 16 + ((lane >> 4) << 3) + r;
        int  n = col0 + wn * 64 + nt * 16 + (lane & 15);
        float v = acc[mt][nt][r] + bias[n];
        if (pos)   v += pos[(int)(m % SS) * N + n];
        if (resid) v += resid[m * (long)N + n];
        if (relu && v < 0.0f) v = 0.0f;
        if (Cf) Cf[m * (long)N + n] = v;
        if (Ch) Ch[m * (long)N + n] = (_Float16)v;
      }
    }
  }
}

// ---------------------------------------------------------------------------
// Fused per-(sequence, head) attention.
//   grid = (1536, 12), block = 128 (4 waves).
//   S=36 padded to 48 rows; HD=64. All GEMMs via WMMA f16 (K steps of 32).
//   z += O written directly (each head owns a disjoint 64-col slice).
// ---------------------------------------------------------------------------
__global__ __launch_bounds__(128) void k_attn(const _Float16* __restrict__ hb,
                                              const _Float16* __restrict__ wqkv,
                                              const float* __restrict__ bq,
                                              const float* __restrict__ bk,
                                              const float* __restrict__ bv,
                                              float* __restrict__ zf) {
  const int bc = blockIdx.x;
  const int h  = blockIdx.y;
  __shared__ __align__(16) char smem[51200];
  _Float16* Xs  = (_Float16*)(smem + 0);      // [48][64] LN1 slice (pad rows 0)
  _Float16* Wsm = (_Float16*)(smem + 6144);   // 3 x [64][64] Wq,Wk,Wv
  float*    Sc  = (float*)(smem + 0);         // [48][48] scores (overlaps Xs/W)
  _Float16* Ps  = (_Float16*)(smem + 9216);   // [48][64] probs, K-padded
  _Float16* Qs  = (_Float16*)(smem + 30720);  // [48][64]
  _Float16* Ks  = (_Float16*)(smem + 36864);  // [48][64]
  _Float16* VT  = (_Float16*)(smem + 43008);  // [64][64] V transposed (n, k)

  int tid = threadIdx.x, wave = tid >> 5, lane = tid & 31;

  for (int i = tid; i < 48 * 64; i += 128) {
    int r = i >> 6, c = i & 63;
    Xs[i] = (r < SS) ? hb[((long)bc * SS + r) * DD + h * HDD + c] : (_Float16)0.0f;
  }
  for (int i = tid; i < 3 * 4096; i += 128) {
    int mat = i >> 12, r = i & 4095;
    Wsm[i] = wqkv[((long)(mat * HH + h)) * 4096 + r];
  }
  for (int i = tid; i < 64 * 64; i += 128) VT[i] = (_Float16)0.0f;
  __syncthreads();

  // Q,K,V = X @ W^T + bias   (36 tile jobs over 4 waves)
  for (int j = wave; j < 36; j += 4) {
    int mat = j / 12, t = j % 12, mt = t >> 2, nt = t & 3;
    const _Float16* Wm = Wsm + mat * 4096;
    v8f acc;
#pragma unroll
    for (int i = 0; i < 8; ++i) acc[i] = 0.0f;
#pragma unroll
    for (int kk = 0; kk < 64; kk += 32) {
      v16h a = lda_op(Xs, mt * 16, kk, 64, lane);
      v16h b = ldb_op(Wm, nt * 16, kk, 64, lane);
      acc = wmma32(a, b, acc);
    }
    const float* bias = ((mat == 0) ? bq : (mat == 1) ? bk : bv) + h * HDD;
#pragma unroll
    for (int r = 0; r < 8; ++r) {
      int m = mt * 16 + ((lane >> 4) << 3) + r;
      int n = nt * 16 + (lane & 15);
      float v = acc[r] + bias[n];
      if (mat == 0)      Qs[m * 64 + n] = (_Float16)v;
      else if (mat == 1) Ks[m * 64 + n] = (_Float16)v;
      else               VT[n * 64 + m] = (_Float16)v;
    }
  }
  __syncthreads();

  // scores = (Q @ K^T) * 1/8   (9 tile jobs)
  for (int j = wave; j < 9; j += 4) {
    int mt = j / 3, nt = j % 3;
    v8f acc;
#pragma unroll
    for (int i = 0; i < 8; ++i) acc[i] = 0.0f;
#pragma unroll
    for (int kk = 0; kk < 64; kk += 32) {
      v16h a = lda_op(Qs, mt * 16, kk, 64, lane);
      v16h b = ldb_op(Ks, nt * 16, kk, 64, lane);   // B[k][n] = K[n][k]
      acc = wmma32(a, b, acc);
    }
#pragma unroll
    for (int r = 0; r < 8; ++r) {
      int m = mt * 16 + ((lane >> 4) << 3) + r;
      int n = nt * 16 + (lane & 15);
      Sc[m * 48 + n] = acc[r] * 0.125f;
    }
  }
  __syncthreads();

  // row softmax (rows 0..35 over cols 0..35); write f16 probs K-padded
  if (tid < SS) {
    float mx = -1e30f;
    for (int c = 0; c < SS; ++c) { float s = Sc[tid * 48 + c]; mx = s > mx ? s : mx; }
    float e[SS], sum = 0.0f;
    for (int c = 0; c < SS; ++c) { e[c] = __expf(Sc[tid * 48 + c] - mx); sum += e[c]; }
    float inv = 1.0f / sum;
    for (int c = 0; c < SS; ++c) Ps[tid * 64 + c] = (_Float16)(e[c] * inv);
    for (int c = SS; c < 64; ++c) Ps[tid * 64 + c] = (_Float16)0.0f;
  }
  for (int i = tid; i < 12 * 64; i += 128) Ps[SS * 64 + i] = (_Float16)0.0f;
  __syncthreads();

  // O = P @ V ; z += O   (12 tile jobs)
  for (int j = wave; j < 12; j += 4) {
    int mt = j >> 2, nt = j & 3;
    v8f acc;
#pragma unroll
    for (int i = 0; i < 8; ++i) acc[i] = 0.0f;
#pragma unroll
    for (int kk = 0; kk < 64; kk += 32) {
      v16h a = lda_op(Ps, mt * 16, kk, 64, lane);
      v16h b = ldb_op(VT, nt * 16, kk, 64, lane);   // B[k][n] = V[k][n] = VT[n][k]
      acc = wmma32(a, b, acc);
    }
#pragma unroll
    for (int r = 0; r < 8; ++r) {
      int m = mt * 16 + ((lane >> 4) << 3) + r;
      if (m < SS) {
        int n = nt * 16 + (lane & 15);
        long idx = ((long)bc * SS + m) * DD + h * HDD + n;
        zf[idx] += acc[r];
      }
    }
  }
}

// ---------------------------------------------------------------------------
// Host launcher
// ---------------------------------------------------------------------------
extern "C" void kernel_launch(void* const* d_in, const int* in_sizes, int n_in,
                              void* d_out, int out_size, void* d_ws, size_t ws_size,
                              hipStream_t stream) {
  const float* batch   = (const float*)d_in[0];
  const float* W_patch = (const float*)d_in[1];
  const float* b_patch = (const float*)d_in[2];
  const float* pos     = (const float*)d_in[3];
  const float* ln1_g   = (const float*)d_in[4];
  const float* ln1_b   = (const float*)d_in[5];
  const float* Wq      = (const float*)d_in[6];
  const float* bq      = (const float*)d_in[7];
  const float* Wk      = (const float*)d_in[8];
  const float* bk      = (const float*)d_in[9];
  const float* Wv      = (const float*)d_in[10];
  const float* bv      = (const float*)d_in[11];
  const float* ln2_g   = (const float*)d_in[12];
  const float* ln2_b   = (const float*)d_in[13];
  const float* W1      = (const float*)d_in[14];
  const float* b1      = (const float*)d_in[15];
  const float* W2      = (const float*)d_in[16];
  const float* b2      = (const float*)d_in[17];
  const float* W_out   = (const float*)d_in[18];
  const float* b_out   = (const float*)d_in[19];

  char* ws = (char*)d_ws;
  float*    zf     = (float*)   (ws + 0);          // 55296*768 f32
  _Float16* zb     = (_Float16*)(ws + 169869312);  // 55296*768 f16
  _Float16* hb     = (_Float16*)(ws + 254803968);  // 55296*768 f16 (LN1 out)
  _Float16* hidb   = (_Float16*)(ws + 339738624);  // 55296*1536 f16
  _Float16* Ap     = (_Float16*)(ws + 509607936);  // 55296*64 f16
  _Float16* Wph    = (_Float16*)(ws + 516685824);  // 768*64
  _Float16* Wqkvh  = (_Float16*)(ws + 516784128);  // L*3*H*4096
  _Float16* W1h    = (_Float16*)(ws + 517963776);  // L*1536*768
  _Float16* W2h    = (_Float16*)(ws + 527400960);  // L*768*1536
  _Float16* Wouth  = (_Float16*)(ws + 536838144);  // 256*768

  // ---- weight conversion (once per launch) ----
  k_cvt_wpatch<<<(DD * 64 + 255) / 256, 256, 0, stream>>>(W_patch, Wph);
  k_cvt_qkv<<<(LL * HH * 4096 + 255) / 256, 256, 0, stream>>>(Wq, Wqkvh, 0);
  k_cvt_qkv<<<(LL * HH * 4096 + 255) / 256, 256, 0, stream>>>(Wk, Wqkvh, 1);
  k_cvt_qkv<<<(LL * HH * 4096 + 255) / 256, 256, 0, stream>>>(Wv, Wqkvh, 2);
  k_cvt_f16<<<(LL * 1536 * 768 + 255) / 256, 256, 0, stream>>>(W1, W1h, LL * 1536 * 768);
  k_cvt_f16<<<(LL * 768 * 1536 + 255) / 256, 256, 0, stream>>>(W2, W2h, LL * 768 * 1536);
  k_cvt_f16<<<(OUTN * DD + 255) / 256, 256, 0, stream>>>(W_out, Wouth, OUTN * DD);

  // ---- patch embedding: z = patches @ W_patch^T + b_patch + pos ----
  k_patchify<<<TT, 64, 0, stream>>>(batch, Ap);
  k_gemm<<<dim3(TT / 128, DD / 128), 256, 0, stream>>>(
      Ap, Wph, b_patch, pos, nullptr, zf, nullptr, DD, 64, 0);

  // ---- transformer layers ----
  for (int l = 0; l < LL; ++l) {
    k_ln<<<TT, 256, 0, stream>>>(zf, ln1_g + l * DD, ln1_b + l * DD, nullptr, hb);
    k_attn<<<dim3(BB * CC, HH), 128, 0, stream>>>(
        hb, Wqkvh + (long)l * 3 * HH * 4096,
        bq + l * HH * HDD, bk + l * HH * HDD, bv + l * HH * HDD, zf);
    k_ln<<<TT, 256, 0, stream>>>(zf, ln2_g + l * DD, ln2_b + l * DD, zf, zb);
    k_gemm<<<dim3(TT / 128, 1536 / 128), 256, 0, stream>>>(
        zb, W1h + (long)l * 1536 * 768, b1 + l * 1536, nullptr, nullptr,
        nullptr, hidb, 1536, 768, 1);
    k_gemm<<<dim3(TT / 128, DD / 128), 256, 0, stream>>>(
        hidb, W2h + (long)l * 768 * 1536, b2 + l * DD, nullptr, zf,
        zf, zb, DD, 1536, 0);
  }

  // ---- output head: out = z @ W_out^T + b_out ----
  k_gemm<<<dim3(TT / 128, OUTN / 128), 256, 0, stream>>>(
      zb, Wouth, b_out, nullptr, nullptr, (float*)d_out, nullptr, OUTN, DD, 0);

  (void)in_sizes; (void)n_in; (void)out_size; (void)ws_size;
}
